// KANLinear_6287832121478
// MI455X (gfx1250) — matrix operand order repaired and checked by reading, not compiled, not run on hardware
//
#include <hip/hip_runtime.h>
#include <hip/hip_bf16.h>

// ---- problem constants (from reference) ----
#define IN_F   1024
#define OUT_F  64
#define NCOEF  8          // GRID_SIZE + SPLINE_ORDER
#define BATCH  8192
#define GRID_H 0.4f       // uniform knot spacing on [-1,1], 5 intervals

typedef __attribute__((ext_vector_type(16))) __bf16       v16bf;
typedef __attribute__((ext_vector_type(8)))  float        v8f;
typedef __attribute__((ext_vector_type(4)))  unsigned int v4u;

__device__ __forceinline__ unsigned short f2bf(float f) {
    unsigned int u = __float_as_uint(f);
    u += 0x7FFFu + ((u >> 16) & 1u);      // round-to-nearest-even
    return (unsigned short)(u >> 16);
}

// -----------------------------------------------------------------------------
// Preprocess: fold base matmul into spline coefficients via Greville abscissae
//   coef'[i][o][k] = bf16( coef[i,o,k] + xi_k * base_weight[i,o] )
//   xi_k = -1 + (k-1)*h  (cubic B-splines reproduce linear functions exactly)
// -----------------------------------------------------------------------------
__global__ void kan_fold_coef(const float* __restrict__ coef,
                              const float* __restrict__ w,
                              unsigned short* __restrict__ coef_bf) {
    int idx = blockIdx.x * blockDim.x + threadIdx.x;    // over IN_F*OUT_F*NCOEF
    if (idx >= IN_F * OUT_F * NCOEF) return;
    int k  = idx & 7;
    int io = idx >> 3;
    float xi = -1.0f + (float)(k - 1) * GRID_H;
    coef_bf[idx] = f2bf(coef[idx] + xi * w[io]);
}

// -----------------------------------------------------------------------------
// Main fused kernel: generate cubic B-spline bases on the fly, GEMM via WMMA.
//   out[b,o] = bias[o] + sum_{i,k} bases[b,i,k] * coef'[i,o,k]
// Each wave: 16 batch rows x 64 outputs (4x v8f accumulators).
// -----------------------------------------------------------------------------
#define ROWS_PW   16
#define WAVES_PB  4
#define ROW_STRIDE 528                  // 32 inputs * 8 bf16 = 512B + 16B pad
#define WAVE_LDS  (ROWS_PW * ROW_STRIDE)

__global__ __launch_bounds__(128)
void kan_main(const float* __restrict__ x,
              const unsigned short* __restrict__ coef_bf,
              const float* __restrict__ bias,
              float* __restrict__ out) {
    __shared__ __align__(16) unsigned char lds[WAVES_PB * WAVE_LDS];

    const int lane = threadIdx.x & 31;
    const int wave = threadIdx.x >> 5;
    const int half = lane >> 4;           // 0: lanes 0-15, 1: lanes 16-31
    const int l16  = lane & 15;
    const int row_base = blockIdx.x * (WAVES_PB * ROWS_PW) + wave * ROWS_PW;

    unsigned char* wl = &lds[wave * WAVE_LDS];   // wave-private LDS tile

    v8f acc[4] = {};                      // 4 N-tiles of 16 outputs

    for (int ic = 0; ic < IN_F / 32; ++ic) {
        const int ib = ic * 32;

        // ---- phase 1: bases for 16 rows x 32 inputs (lane owns input ib+lane)
        #pragma unroll 4
        for (int r = 0; r < ROWS_PW; ++r) {
            const float* xp = &x[(size_t)(row_base + r) * IN_F + ib + lane];
            float xv = *xp;
            if (ic + 1 < IN_F / 32)
                __builtin_prefetch(xp + 32, 0, 0);   // next chunk -> global_prefetch

            // uniform cubic B-spline: cell + 4 nonzero blending weights
            float t  = (xv + 1.0f) * 2.5f;           // (x+1)/h
            float jf = fminf(fmaxf(floorf(t), 0.0f), 4.0f);
            int   j  = (int)jf;
            float u  = t - jf;
            float u2 = u * u, u3 = u2 * u;
            float um = 1.0f - u;
            unsigned short wh[4];
            wh[0] = f2bf(um * um * um * (1.0f / 6.0f));
            wh[1] = f2bf((3.0f * u3 - 6.0f * u2 + 4.0f) * (1.0f / 6.0f));
            wh[2] = f2bf((-3.0f * u3 + 3.0f * u2 + 3.0f * u + 1.0f) * (1.0f / 6.0f));
            wh[3] = f2bf(u3 * (1.0f / 6.0f));

            // scatter into 8-wide K slice (indices j..j+3 nonzero), pack b128
            unsigned int pk[4];
            #pragma unroll
            for (int q = 0; q < 4; ++q) {
                int m0 = 2 * q, m1 = 2 * q + 1;
                unsigned int lo = (m0 >= j && m0 < j + 4) ? (unsigned int)wh[m0 - j] : 0u;
                unsigned int hi = (m1 >= j && m1 < j + 4) ? (unsigned int)wh[m1 - j] : 0u;
                pk[q] = lo | (hi << 16);
            }
            v4u pv = { pk[0], pk[1], pk[2], pk[3] };
            *(v4u*)(wl + r * ROW_STRIDE + lane * 16) = pv;   // ds_store_b128
        }
        __syncthreads();

        // ---- phase 2: 8 K-steps of 32 x 4 N-tiles = 32 WMMA per chunk
        #pragma unroll
        for (int t = 0; t < 8; ++t) {
            // A fragment 16x32 bf16 (ISA layout: lanes 0-15 K lo-half, 16-31 hi-half)
            union { v16bf h; v4u q[2]; } A;
            const unsigned char* arow = wl + l16 * ROW_STRIDE + t * 64 + half * 16;
            A.q[0] = *(const v4u*)(arow);        // K = t*32 + {0..7 | 8..15}
            A.q[1] = *(const v4u*)(arow + 32);   // K = t*32 + {16..23 | 24..31}

            #pragma unroll
            for (int nt = 0; nt < 4; ++nt) {
                // B fragment 32x16 bf16: column = lane%16, K rows from coef'
                union { v16bf h; v4u q[2]; } B;
                int o  = nt * 16 + l16;
                int i0 = ib + t * 4 + half;
                B.q[0] = *(const v4u*)(coef_bf + ((size_t)i0 * OUT_F + o) * NCOEF);
                B.q[1] = *(const v4u*)(coef_bf + ((size_t)(i0 + 2) * OUT_F + o) * NCOEF);

                acc[nt] = __builtin_amdgcn_wmma_f32_16x16x32_bf16(
                    false, A.h, false, B.h, (short)0, acc[nt], false, false);
            }
        }
        __syncthreads();
    }

    // ---- epilogue: bias + store (C layout: vgpr r -> M = r + half*8, N = lane%16)
    #pragma unroll
    for (int nt = 0; nt < 4; ++nt) {
        int col = nt * 16 + l16;
        float bv = bias[col];
        #pragma unroll
        for (int r = 0; r < 8; ++r) {
            int row = row_base + r + half * 8;
            out[(size_t)row * OUT_F + col] = acc[nt][r] + bv;
        }
    }
}

// -----------------------------------------------------------------------------
extern "C" void kernel_launch(void* const* d_in, const int* in_sizes, int n_in,
                              void* d_out, int out_size, void* d_ws, size_t ws_size,
                              hipStream_t stream) {
    const float* x    = (const float*)d_in[0];
    const float* grid = (const float*)d_in[1];   // uniform linspace; closed-form used
    const float* coef = (const float*)d_in[2];
    const float* bw   = (const float*)d_in[3];
    const float* bias = (const float*)d_in[4];
    (void)grid; (void)in_sizes; (void)n_in; (void)out_size; (void)ws_size;

    unsigned short* coef_bf = (unsigned short*)d_ws;   // 1 MB scratch

    kan_fold_coef<<<(IN_F * OUT_F * NCOEF + 255) / 256, 256, 0, stream>>>(
        coef, bw, coef_bf);

    kan_main<<<BATCH / (WAVES_PB * ROWS_PW), WAVES_PB * 32, 0, stream>>>(
        x, coef_bf, bias, (float*)d_out);
}
